// GraphWalkerMemory_46918222741860
// MI455X (gfx1250) — matrix operand order repaired and checked by reading, not compiled, not run on hardware
//
#include <hip/hip_runtime.h>
#include <math.h>

// ---------------- problem constants ----------------
#define Bn   16
#define Hn   16
#define Nn   4096
#define Tn   8
#define DS   256
#define DID  128
#define Ln   8
#define Kn   32
#define HSn  8
#define DQn  32
#define DHID 1024
#define DST  896          // 3*DS + DID
#define QH   512          // 2*HS*DQ
#define QO   256          // HS*DQ
#define NW   (Bn*Hn)      // 256 walkers
#define NGRP 24           // max plane groups of 16

typedef __attribute__((ext_vector_type(16))) _Float16 v16h;
typedef __attribute__((ext_vector_type(8)))  float    v8f;

// ---------------- WMMA helpers (ISA 7.12.2 layouts) ----------------
__device__ __forceinline__ v8f wmma16(v16h a, v16h b, v8f c) {
  return __builtin_amdgcn_wmma_f32_16x16x32_f16(false, a, false, b, (short)0, c,
                                                false, false);
}

// A fragment: 16x32 f16 tile from LDS, row-major with element stride `stride`.
// lane<16: row=lane, K={0..7,16..23}; lane>=16: row=lane-16, K={8..15,24..31}
__device__ __forceinline__ v16h lds_load_A(const _Float16* base, int stride, int kBase) {
  const int lane = threadIdx.x & 31;
  const int m    = lane & 15;
  const int koff = ((lane >> 4) & 1) << 3;   // 0 or 8
  const _Float16* row = base + m * stride + kBase + koff;
  v16h a;
#pragma unroll
  for (int i = 0; i < 8; ++i) a[i] = row[i];
#pragma unroll
  for (int i = 0; i < 8; ++i) a[8 + i] = row[16 + i];
  return a;
}

// B fragment from pre-swizzled weights: lane's 16 halves are contiguous 32B.
// layout: dst[(((nt*K32)+kt)*32 + lane)*16 + i]
__device__ __forceinline__ v16h gld_load_B_swz(const _Float16* wswz, int K32, int kt, int nt) {
  const int lane = threadIdx.x & 31;
  const _Float16* p = wswz + ((size_t)((nt * K32) + kt) * 32 + lane) * 16;
  return *reinterpret_cast<const v16h*>(p);
}

__device__ __forceinline__ float gelu_exact(float x) {
  return 0.5f * x * (1.0f + erff(x * 0.70710678118654752f));
}

// ---------------- swizzled weight conversion ----------------
// B-tile element for tile (kt,nt): n = nt*16 + (lane&15), k = kt*32 + ((ii>>3)<<4) + ((lane>>4)<<3) + (ii&7)
__device__ __forceinline__ void swz_decode(int j, int K, int N, int& k, int& n) {
  int ii   = j & 15;
  int lane = (j >> 4) & 31;
  int rem  = j >> 9;
  int K32  = K >> 5;
  int kt   = rem % K32;
  int nt   = rem / K32;
  n = nt * 16 + (lane & 15);
  k = kt * 32 + ((ii >> 3) << 4) + ((lane >> 4) << 3) + (ii & 7);
}

// plain weights stored row-major [K][N]: src = w[k*N + n]
__global__ void cvt_swz_plain(const float* __restrict__ src, _Float16* __restrict__ dst,
                              int K, int N) {
  int i = blockIdx.x * 256 + threadIdx.x;
  if (i >= K * N) return;
  int k, n;
  swz_decode(i, K, N, k, n);
  dst[i] = (_Float16)src[(size_t)k * N + n];
}

// per-plane weights stored [P][N][K] (output-major): src = w[((p*N)+n)*K + k]
__global__ void cvt_swz_plane(const float* __restrict__ src, _Float16* __restrict__ dst,
                              int K, int N, int P) {
  int i = blockIdx.x * 256 + threadIdx.x;
  if (i >= P * K * N) return;
  int p = i / (K * N);
  int j = i % (K * N);
  int k, n;
  swz_decode(j, K, N, k, n);
  dst[i] = (_Float16)src[((size_t)p * N + n) * K + k];
}

// ---------------- misc prep ----------------
__global__ void prep_misc(const int* __restrict__ walker_pos, int* curpos, float* mpacc) {
  int i = threadIdx.x;
  if (i < NW) curpos[i] = walker_pos[i];
  if (i < Kn) mpacc[i] = 0.f;
}

// counting-sort walkers into plane-uniform padded groups of 16
__global__ void build_groups(const int* __restrict__ plane_idx,
                             int* __restrict__ groupPlane, int* __restrict__ groupRows) {
  __shared__ int cnt[Ln], off[Ln], cur[Ln];
  int tid = threadIdx.x;
  if (tid < Ln) { cnt[tid] = 0; cur[tid] = 0; }
  for (int i = tid; i < NGRP * 16; i += 256) groupRows[i] = -1;
  __syncthreads();
  int p = plane_idx[tid];
  atomicAdd(&cnt[p], 1);
  __syncthreads();
  if (tid == 0) {
    int g = 0;
    for (int pp = 0; pp < Ln; ++pp) {
      off[pp] = g * 16;
      int ng = (cnt[pp] + 15) >> 4;
      for (int i = 0; i < ng; ++i) groupPlane[g++] = pp;
    }
    for (; g < NGRP; ++g) groupPlane[g] = -1;
  }
  __syncthreads();
  int slot = off[p] + atomicAdd(&cur[p], 1);
  groupRows[slot] = tid;
}

// ---------------- Kall = mlp2(node_id, kw) precompute ----------------
__global__ void kall_kernel(const float* __restrict__ node_id,
                            const _Float16* __restrict__ kw1s, const float* __restrict__ kb1,
                            const _Float16* __restrict__ kw2s, const float* __restrict__ kb2,
                            float* __restrict__ Kall) {
  __shared__ __align__(16) _Float16 sId[16 * 136];
  __shared__ __align__(16) _Float16 sHk[16 * 520];
  const int row0 = blockIdx.x * 16;
  for (int e = threadIdx.x; e < 16 * DID; e += 256) {
    int r = e >> 7, d = e & 127;
    sId[r * 136 + d] = (_Float16)node_id[(size_t)(row0 + r) * DID + d];
  }
  __syncthreads();
  const int wave = threadIdx.x >> 5;
  const int lane = threadIdx.x & 31;
  const int n    = lane & 15;
  const int rb   = ((lane >> 4) & 1) << 3;
  // GEMM1: [16,128] x [128,512] -> gelu -> sHk   (K32 = 4)
  for (int tt = 0; tt < 4; ++tt) {
    int nt = wave * 4 + tt;
    v8f acc = {};
    for (int kc = 0; kc < 4; ++kc)
      acc = wmma16(lds_load_A(sId, 136, kc * 32), gld_load_B_swz(kw1s, 4, kc, nt), acc);
    float bias = kb1[nt * 16 + n];
#pragma unroll
    for (int r = 0; r < 8; ++r)
      sHk[(r + rb) * 520 + nt * 16 + n] = (_Float16)gelu_exact(acc[r] + bias);
  }
  __syncthreads();
  // GEMM2: [16,512] x [512,256] -> Kall   (K32 = 16)
  for (int tt = 0; tt < 2; ++tt) {
    int nt = wave * 2 + tt;
    v8f acc = {};
    for (int kc = 0; kc < 16; ++kc)
      acc = wmma16(lds_load_A(sHk, 520, kc * 32), gld_load_B_swz(kw2s, 16, kc, nt), acc);
    float bias = kb2[nt * 16 + n];
#pragma unroll
    for (int r = 0; r < 8; ++r)
      Kall[(size_t)(row0 + r + rb) * QO + nt * 16 + n] = acc[r] + bias;
  }
}

// ---------------- per-step compute ----------------
__global__ void step_compute(const float* __restrict__ sOut,
                             const float* __restrict__ node_id,
                             const float* __restrict__ token_embed,
                             const float* __restrict__ norm_w,
                             const _Float16* __restrict__ W1s, const float* __restrict__ b1,
                             const _Float16* __restrict__ W2s, const float* __restrict__ b2,
                             const _Float16* __restrict__ qw1s, const float* __restrict__ qb1,
                             const _Float16* __restrict__ qw2s, const float* __restrict__ qb2,
                             const float* __restrict__ Kall,
                             const int* __restrict__ neighbors,
                             const int* __restrict__ curpos,
                             const float* __restrict__ wsOut,
                             const int* __restrict__ groupPlane,
                             const int* __restrict__ groupRows,
                             float* __restrict__ contentBuf,
                             float* __restrict__ mpacc,
                             int* __restrict__ nextpos,
                             int t) {
  const int p = groupPlane[blockIdx.x];
  if (p < 0) return;

  __shared__ __align__(16) _Float16 sSteer[16 * 904];   // reused as q[16][256] f32 later
  __shared__ __align__(16) _Float16 sH[16 * 1032];      // hidden (1024) then qh (512)
  __shared__ int rowIdx[16];

  const int wave = threadIdx.x >> 5;
  const int lane = threadIdx.x & 31;
  if (threadIdx.x < 16) rowIdx[threadIdx.x] = groupRows[blockIdx.x * 16 + threadIdx.x];
  __syncthreads();

  // ---- build steer rows (f16): [rms(s_cur) | id | wstate | tok] ----
  for (int rr = 0; rr < 2; ++rr) {
    int r = wave * 2 + rr;
    int widx = rowIdx[r];
    _Float16* srow16 = sSteer + r * 904;
    if (widx < 0) {
      for (int d = lane; d < DST; d += 32) srow16[d] = (_Float16)0.f;
      continue;
    }
    int b = widx >> 4;
    int pos = curpos[widx];
    const float* scur = sOut + ((size_t)b * Nn + pos) * DS;
    float ss = 0.f;
    for (int d = lane; d < DS; d += 32) { float v = scur[d]; ss += v * v; }
#pragma unroll
    for (int off = 16; off; off >>= 1) ss += __shfl_xor(ss, off, 32);
    float rn = rsqrtf(ss * (1.0f / DS) + 1e-6f);
    for (int d = lane; d < DS; d += 32) srow16[d] = (_Float16)(scur[d] * rn * norm_w[d]);
    const float* idr = node_id + (size_t)pos * DID;
    for (int d = lane; d < DID; d += 32) srow16[DS + d] = (_Float16)idr[d];
    const float* wr = wsOut + (size_t)widx * DS;
    for (int d = lane; d < DS; d += 32) srow16[DS + DID + d] = (_Float16)wr[d];
    const float* tr = token_embed + ((size_t)b * Tn + t) * DS;
    for (int d = lane; d < DS; d += 32) srow16[2 * DS + DID + d] = (_Float16)tr[d];
  }
  __syncthreads();

  const int n  = lane & 15;
  const int rb = ((lane >> 4) & 1) << 3;

  // ---- GEMM1: hidden = gelu(steer @ W1[p]^T + b1[p])  [16,896]x[896,1024], K32=28 ----
  {
    const _Float16* w = W1s + (size_t)p * DST * DHID;
    const float* bias1 = b1 + p * DHID;
    for (int tt = 0; tt < 8; ++tt) {
      int nt = wave * 8 + tt;
      v8f acc = {};
      for (int kc = 0; kc < DST / 32; ++kc)
        acc = wmma16(lds_load_A(sSteer, 904, kc * 32), gld_load_B_swz(w, DST / 32, kc, nt), acc);
      float bv = bias1[nt * 16 + n];
#pragma unroll
      for (int r = 0; r < 8; ++r)
        sH[(r + rb) * 1032 + nt * 16 + n] = (_Float16)gelu_exact(acc[r] + bv);
    }
  }
  __syncthreads();

  // ---- GEMM2: content = hidden @ W2[p]^T + b2[p]  [16,1024]x[1024,256], K32=32 ----
  {
    const _Float16* w = W2s + (size_t)p * DHID * DS;
    const float* bias2 = b2 + p * DS;
    for (int tt = 0; tt < 2; ++tt) {
      int nt = wave * 2 + tt;
      v8f acc = {};
      for (int kc = 0; kc < DHID / 32; ++kc)
        acc = wmma16(lds_load_A(sH, 1032, kc * 32), gld_load_B_swz(w, DHID / 32, kc, nt), acc);
      float bv = bias2[nt * 16 + n];
#pragma unroll
      for (int r = 0; r < 8; ++r) {
        int wi = rowIdx[r + rb];
        if (wi >= 0) contentBuf[(size_t)wi * DS + nt * 16 + n] = acc[r] + bv;
      }
    }
  }
  __syncthreads();

  // ---- qh = gelu(steer @ qw1 + qb1)  [16,896]x[896,512] -> sH, K32=28 ----
  for (int tt = 0; tt < 4; ++tt) {
    int nt = wave * 4 + tt;
    v8f acc = {};
    for (int kc = 0; kc < DST / 32; ++kc)
      acc = wmma16(lds_load_A(sSteer, 904, kc * 32), gld_load_B_swz(qw1s, DST / 32, kc, nt), acc);
    float bv = qb1[nt * 16 + n];
#pragma unroll
    for (int r = 0; r < 8; ++r)
      sH[(r + rb) * 1032 + nt * 16 + n] = (_Float16)gelu_exact(acc[r] + bv);
  }
  __syncthreads();

  // ---- q = qh @ qw2 + qb2  [16,512]x[512,256] -> qS (reuse sSteer as f32), K32=16 ----
  float* qS = reinterpret_cast<float*>(sSteer);
  for (int tt = 0; tt < 2; ++tt) {
    int nt = wave * 2 + tt;
    v8f acc = {};
    for (int kc = 0; kc < QH / 32; ++kc)
      acc = wmma16(lds_load_A(sH, 1032, kc * 32), gld_load_B_swz(qw2s, QH / 32, kc, nt), acc);
    float bv = qb2[nt * 16 + n];
#pragma unroll
    for (int r = 0; r < 8; ++r)
      qS[(r + rb) * QO + nt * 16 + n] = acc[r] + bv;
  }
  __syncthreads();

  // ---- neighbor scoring: lane == neighbor index (K==32==wave size) ----
  const float scale = 0.022097086912079612f;   // 1/(HS*sqrt(DQ))
  for (int rr = 0; rr < 2; ++rr) {
    int r = wave * 2 + rr;
    int widx = rowIdx[r];
    if (widx < 0) continue;
    int pos = curpos[widx];
    int nb  = neighbors[(size_t)pos * Kn + lane];
    const float4* k4 = reinterpret_cast<const float4*>(Kall + (size_t)nb * QO);
    const float4* q4 = reinterpret_cast<const float4*>(qS + r * QO);
    float sc = 0.f;
#pragma unroll 8
    for (int d = 0; d < QO / 4; ++d) {
      float4 a = q4[d], b4 = k4[d];
      sc += a.x * b4.x + a.y * b4.y + a.z * b4.z + a.w * b4.w;
    }
    sc *= scale;
    float mx = sc;
#pragma unroll
    for (int off = 16; off; off >>= 1) mx = fmaxf(mx, __shfl_xor(mx, off, 32));
    float e = expf(sc - mx);
    float se = e;
#pragma unroll
    for (int off = 16; off; off >>= 1) se += __shfl_xor(se, off, 32);
    atomicAdd(&mpacc[lane], e / se);
    // argmax (first index on ties, matching jnp.argmax)
    float best = sc; int bi = lane;
#pragma unroll
    for (int off = 16; off; off >>= 1) {
      float ob = __shfl_xor(best, off, 32);
      int   oi = __shfl_xor(bi, off, 32);
      if (ob > best || (ob == best && oi < bi)) { best = ob; bi = oi; }
    }
    if (lane == 0) nextpos[widx] = neighbors[(size_t)pos * Kn + bi];
  }
}

// ---------------- per-step apply (scatter) ----------------
__global__ void step_apply(const float* __restrict__ contentBuf,
                           const int* __restrict__ nextposBuf,
                           int* __restrict__ curpos,
                           float* __restrict__ sOut, float* __restrict__ wsOut,
                           float* __restrict__ posOut,
                           float* __restrict__ co, float* __restrict__ vc) {
  int widx = blockIdx.x;
  int d = threadIdx.x;
  int b = widx >> 4;
  int pos = curpos[widx];
  int np  = nextposBuf[widx];
  float c = contentBuf[(size_t)widx * DS + d];
  atomicAdd(&sOut[((size_t)b * Nn + pos) * DS + d], c);
  wsOut[(size_t)widx * DS + d] += c;
  __syncthreads();           // all threads read curpos before lane 0 rewrites it
  if (d == 0) {
    atomicAdd(&co[(size_t)pos * Nn + np], 1.0f);
    atomicAdd(&vc[np], 1.0f);
    curpos[widx] = np;
    posOut[widx] = (float)np;
  }
}

// ---------------- motor + load-balance finalize ----------------
__global__ void motor_lb(const float* __restrict__ wsOut, float* __restrict__ motor,
                         float* __restrict__ mpacc, float* __restrict__ lb, int t) {
  int b = blockIdx.x, d = threadIdx.x;
  float acc = 0.f;
#pragma unroll
  for (int h = 0; h < Hn; ++h) acc += wsOut[((size_t)b * Hn + h) * DS + d];
  motor[((size_t)b * Tn + t) * DS + d] = acc * (1.0f / Hn);
  if (b == 0 && d == 0) {
    float s = 0.f;
    for (int k = 0; k < Kn; ++k) {
      float m = mpacc[k] * (1.0f / NW);
      s += m * m;
      mpacc[k] = 0.f;
    }
    lb[0] += (float)Kn * s;
  }
}

// ---------------- host launch ----------------
extern "C" void kernel_launch(void* const* d_in, const int* in_sizes, int n_in,
                              void* d_out, int out_size, void* d_ws, size_t ws_size,
                              hipStream_t stream) {
  const float* s           = (const float*)d_in[0];
  const float* node_id     = (const float*)d_in[1];
  const float* walker_st   = (const float*)d_in[2];
  const float* token_embed = (const float*)d_in[3];
  const float* norm_w      = (const float*)d_in[4];
  const float* W1          = (const float*)d_in[5];
  const float* b1          = (const float*)d_in[6];
  const float* W2          = (const float*)d_in[7];
  const float* b2          = (const float*)d_in[8];
  const float* qw1         = (const float*)d_in[9];
  const float* qb1         = (const float*)d_in[10];
  const float* qw2         = (const float*)d_in[11];
  const float* qb2         = (const float*)d_in[12];
  const float* kw1         = (const float*)d_in[13];
  const float* kb1         = (const float*)d_in[14];
  const float* kw2         = (const float*)d_in[15];
  const float* kb2         = (const float*)d_in[16];
  const int*   walker_pos  = (const int*)d_in[17];
  const int*   plane_idx   = (const int*)d_in[18];
  const int*   neighbors   = (const int*)d_in[19];

  // output layout (floats): motor | s_new | pos | ws_new | co | vc | lb
  float* out   = (float*)d_out;
  float* motor = out;                                         // 16*8*256
  float* sOut  = out + 32768;                                 // 16*4096*256
  float* posO  = out + 32768 + 16777216;                      // 256
  float* wsOut = posO + 256;                                  // 16*16*256
  float* co    = wsOut + 65536;                               // 4096*4096
  float* vc    = co + 16777216;                               // 4096
  float* lb    = vc + 4096;                                   // 1

  // workspace layout (swizzled f16 weights + scratch)
  char* w = (char*)d_ws;
  _Float16* W1s  = (_Float16*)(w + 0);                        // 8*896*1024
  _Float16* W2s  = (_Float16*)(w + 14680064);                 // 8*1024*256
  _Float16* qw1s = (_Float16*)(w + 18874368);                 // 896*512
  _Float16* qw2s = (_Float16*)(w + 19791872);                 // 512*256
  _Float16* kw1s = (_Float16*)(w + 20054016);                 // 128*512
  _Float16* kw2s = (_Float16*)(w + 20185088);                 // 512*256
  float*    Kall = (float*)   (w + 20447232);                 // 4096*256 f32
  float*    cBuf = (float*)   (w + 24641536);                 // 256*256 f32
  int*      curp = (int*)     (w + 24903680);                 // 256
  int*      nxtp = (int*)     (w + 24904704);                 // 256
  float*    mpac = (float*)   (w + 24905728);                 // 32
  int*      gPl  = (int*)     (w + 24905856);                 // 24
  int*      gRo  = (int*)     (w + 24905984);                 // 24*16

  // init output state
  hipMemcpyAsync(sOut, s, (size_t)Bn * Nn * DS * sizeof(float),
                 hipMemcpyDeviceToDevice, stream);
  hipMemcpyAsync(wsOut, walker_st, (size_t)NW * DS * sizeof(float),
                 hipMemcpyDeviceToDevice, stream);
  hipMemsetAsync(co, 0, (size_t)Nn * Nn * sizeof(float), stream);
  hipMemsetAsync(vc, 0, (size_t)Nn * sizeof(float), stream);
  hipMemsetAsync(lb, 0, sizeof(float), stream);

  // weight conversion into swizzled fragment-major f16
  cvt_swz_plane<<<(Ln * DST * DHID + 255) / 256, 256, 0, stream>>>(W1, W1s, DST, DHID, Ln);
  cvt_swz_plane<<<(Ln * DHID * DS + 255) / 256, 256, 0, stream>>>(W2, W2s, DHID, DS, Ln);
  cvt_swz_plain<<<(DST * QH + 255) / 256, 256, 0, stream>>>(qw1, qw1s, DST, QH);
  cvt_swz_plain<<<(QH * QO + 255) / 256, 256, 0, stream>>>(qw2, qw2s, QH, QO);
  cvt_swz_plain<<<(DID * QH + 255) / 256, 256, 0, stream>>>(kw1, kw1s, DID, QH);
  cvt_swz_plain<<<(QH * QO + 255) / 256, 256, 0, stream>>>(kw2, kw2s, QH, QO);

  prep_misc<<<1, 256, 0, stream>>>(walker_pos, curp, mpac);
  build_groups<<<1, 256, 0, stream>>>(plane_idx, gPl, gRo);
  kall_kernel<<<Nn / 16, 256, 0, stream>>>(node_id, kw1s, kb1, kw2s, kb2, Kall);

  for (int t = 0; t < Tn; ++t) {
    step_compute<<<NGRP, 256, 0, stream>>>(sOut, node_id, token_embed, norm_w,
                                           W1s, b1, W2s, b2, qw1s, qb1, qw2s, qb2,
                                           Kall, neighbors, curp, wsOut,
                                           gPl, gRo, cBuf, mpac, nxtp, t);
    step_apply<<<NW, 256, 0, stream>>>(cBuf, nxtp, curp, sOut, wsOut, posO, co, vc);
    motor_lb<<<Bn, 256, 0, stream>>>(wsOut, motor, mpac, lb, t);
  }
}
